// Block_73744588472675
// MI455X (gfx1250) — matrix operand order repaired and checked by reading, not compiled, not run on hardware
//
#include <hip/hip_runtime.h>
#include <hip/hip_bf16.h>

// ---------------------------------------------------------------------------
// BEiT-style transformer block for MI455X (gfx1250, wave32, WMMA).
// GEMMs use v_wmma_f32_16x16x32_bf16 with register blocking:
//   big kernel : 2x4 frags/wave (8 WMMAs per 6 fragment loads per K-step)
//   small kernel: 2x2 frags/wave (4 WMMAs per 4 fragment loads per K-step)
// ---------------------------------------------------------------------------

typedef unsigned short u16;
typedef __bf16  bf16x16 __attribute__((ext_vector_type(16)));
typedef float   f32x8   __attribute__((ext_vector_type(8)));

#define BATCH   4
#define SEQ     677
#define HEADS   16
#define DH      64
#define DIM     1024
#define FFN_DIM 4096
#define NPAD    768            // 677 padded: multiple of 256 (big N-tiles) and 128 (small M-tiles)
#define MTOK    (BATCH*SEQ)    // 2708
#define MPAD    2752           // tokens padded to multiple of 64
#define BH      (BATCH*HEADS)  // 64

static __device__ __forceinline__ u16 f2bf(float f) {
    unsigned u = __float_as_uint(f);
    unsigned r = u + 0x7FFFu + ((u >> 16) & 1u);   // round-to-nearest-even
    return (u16)(r >> 16);
}

// A fragment: 16x32 bf16 tile, row-major source, leading dim `ld` (elements).
// Lane L<16 : row M=L,    K = {0..7, 16..23}
// Lane L>=16: row M=L-16, K = {8..15, 24..31}      (ISA 7.12.2, 16-bit A 16x32)
static __device__ __forceinline__ bf16x16 load_fragA(const u16* __restrict__ p, int ld, int lane) {
    int row = lane & 15;
    int kb  = (lane >> 4) << 3;       // 0 or 8
    const u16* q = p + (long long)row * ld + kb;
    union { uint4 u[2]; bf16x16 v; } U;
    U.u[0] = *(const uint4*)(q);      // K = kb .. kb+7
    U.u[1] = *(const uint4*)(q + 16); // K = kb+16 .. kb+23
    return U.v;
}

// B fragment: 32x16 bf16 tile of B = W^T, loaded from row-major W[N,K].
// Lane L<16 : col N=L,    K = {0..7, 16..23} ; Lane L>=16: col N=L-16, K = {8..15, 24..31}
static __device__ __forceinline__ bf16x16 load_fragB(const u16* __restrict__ wrow, int ld, int lane) {
    int n  = lane & 15;
    int kb = (lane >> 4) << 3;        // 0 or 8
    const u16* q = wrow + (long long)n * ld + kb;
    union { uint4 u[2]; bf16x16 v; } U;
    U.u[0] = *(const uint4*)(q);
    U.u[1] = *(const uint4*)(q + 16);
    return U.v;
}

// ---------------------------------------------------------------------------
// Big-tile batched GEMM: C[z] = A[z] (MxK) * W[z] (NxK)^T (+bias)(+GELU).
// 8 waves arranged 2(M) x 4(N); each wave computes 32x64 (2x4 frags, 8 accs).
// Block tile 64(M) x 256(N). Requires M%64==0, Nn%256==0, K%32==0.
// ---------------------------------------------------------------------------
template<bool BF16OUT, bool GELU>
__global__ __launch_bounds__(256)
void k_gemm_big(const u16* __restrict__ A, const u16* __restrict__ W,
                const float* __restrict__ bias, void* __restrict__ Cout,
                int K, int lda, int ldw, int ldc,
                long long sA, long long sW, long long sC)
{
    int z = blockIdx.z;
    A += (long long)z * sA;
    W += (long long)z * sW;
    long long coff = (long long)z * sC;

    int lane = threadIdx.x & 31;
    int wid  = threadIdx.x >> 5;
    int wm   = wid & 1;               // 0..1
    int wn   = wid >> 1;              // 0..3
    int m0 = blockIdx.x * 64  + wm * 32;
    int n0 = blockIdx.y * 256 + wn * 64;

    f32x8 acc[2][4];
#pragma unroll
    for (int i = 0; i < 2; ++i)
#pragma unroll
        for (int j = 0; j < 4; ++j)
            acc[i][j] = (f32x8){0.f, 0.f, 0.f, 0.f, 0.f, 0.f, 0.f, 0.f};

    const u16* arow0 = A + (long long)(m0     ) * lda;
    const u16* arow1 = A + (long long)(m0 + 16) * lda;
    const u16* wrow  = W + (long long)n0 * ldw;

    for (int k = 0; k < K; k += 32) {
        bf16x16 a0 = load_fragA(arow0 + k, lda, lane);
        bf16x16 a1 = load_fragA(arow1 + k, lda, lane);
        bf16x16 b0 = load_fragB(wrow + k,                         ldw, lane);
        bf16x16 b1 = load_fragB(wrow + (long long)16 * ldw + k,   ldw, lane);
        bf16x16 b2 = load_fragB(wrow + (long long)32 * ldw + k,   ldw, lane);
        bf16x16 b3 = load_fragB(wrow + (long long)48 * ldw + k,   ldw, lane);
        acc[0][0] = __builtin_amdgcn_wmma_f32_16x16x32_bf16(false, a0, false, b0, (short)0, acc[0][0], false, false);
        acc[0][1] = __builtin_amdgcn_wmma_f32_16x16x32_bf16(false, a0, false, b1, (short)0, acc[0][1], false, false);
        acc[0][2] = __builtin_amdgcn_wmma_f32_16x16x32_bf16(false, a0, false, b2, (short)0, acc[0][2], false, false);
        acc[0][3] = __builtin_amdgcn_wmma_f32_16x16x32_bf16(false, a0, false, b3, (short)0, acc[0][3], false, false);
        acc[1][0] = __builtin_amdgcn_wmma_f32_16x16x32_bf16(false, a1, false, b0, (short)0, acc[1][0], false, false);
        acc[1][1] = __builtin_amdgcn_wmma_f32_16x16x32_bf16(false, a1, false, b1, (short)0, acc[1][1], false, false);
        acc[1][2] = __builtin_amdgcn_wmma_f32_16x16x32_bf16(false, a1, false, b2, (short)0, acc[1][2], false, false);
        acc[1][3] = __builtin_amdgcn_wmma_f32_16x16x32_bf16(false, a1, false, b3, (short)0, acc[1][3], false, false);
    }

    // C/D layout: lane 0-15 -> N=lane, M=r ; lane 16-31 -> N=lane-16, M=8+r
    int nl = lane & 15;
    int mh = (lane >> 4) << 3;
#pragma unroll
    for (int i = 0; i < 2; ++i) {
#pragma unroll
        for (int j = 0; j < 4; ++j) {
            int n  = n0 + j * 16 + nl;
            int mb = m0 + i * 16 + mh;
            float bv = bias ? bias[n] : 0.f;
#pragma unroll
            for (int r = 0; r < 8; ++r) {
                float v = acc[i][j][r] + bv;
                if (GELU) v = 0.5f * v * (1.0f + erff(v * 0.70710678118f));
                long long idx = coff + (long long)(mb + r) * ldc + n;
                if (BF16OUT) ((u16*)Cout)[idx]   = f2bf(v);
                else         ((float*)Cout)[idx] = v;
            }
        }
    }
}

// ---------------------------------------------------------------------------
// Small-N batched GEMM for attn-out (Nn=64): 8 waves arranged 4(M) x 2(N),
// each wave computes 32x32 (2x2 frags, 4 accs). Block tile 128(M) x 64(N).
// Requires M%128==0, Nn%64==0, K%32==0.
// C per z: C + (z/zdiv)*sC0 + (z%zdiv)*sC1.
// ---------------------------------------------------------------------------
template<bool BF16OUT>
__global__ __launch_bounds__(256)
void k_gemm_small(const u16* __restrict__ A, const u16* __restrict__ W,
                  void* __restrict__ Cout,
                  int K, int lda, int ldw, int ldc,
                  long long sA, long long sW, int zdiv, long long sC0, long long sC1)
{
    int z = blockIdx.z;
    A += (long long)z * sA;
    W += (long long)z * sW;
    long long coff = (long long)(z / zdiv) * sC0 + (long long)(z % zdiv) * sC1;

    int lane = threadIdx.x & 31;
    int wid  = threadIdx.x >> 5;
    int wm   = wid >> 1;              // 0..3
    int wn   = wid & 1;               // 0..1
    int m0 = blockIdx.x * 128 + wm * 32;
    int n0 = blockIdx.y * 64  + wn * 32;

    f32x8 acc[2][2];
#pragma unroll
    for (int i = 0; i < 2; ++i)
#pragma unroll
        for (int j = 0; j < 2; ++j)
            acc[i][j] = (f32x8){0.f, 0.f, 0.f, 0.f, 0.f, 0.f, 0.f, 0.f};

    const u16* arow0 = A + (long long)(m0     ) * lda;
    const u16* arow1 = A + (long long)(m0 + 16) * lda;
    const u16* wrow  = W + (long long)n0 * ldw;

    for (int k = 0; k < K; k += 32) {
        bf16x16 a0 = load_fragA(arow0 + k, lda, lane);
        bf16x16 a1 = load_fragA(arow1 + k, lda, lane);
        bf16x16 b0 = load_fragB(wrow + k,                       ldw, lane);
        bf16x16 b1 = load_fragB(wrow + (long long)16 * ldw + k, ldw, lane);
        acc[0][0] = __builtin_amdgcn_wmma_f32_16x16x32_bf16(false, a0, false, b0, (short)0, acc[0][0], false, false);
        acc[0][1] = __builtin_amdgcn_wmma_f32_16x16x32_bf16(false, a0, false, b1, (short)0, acc[0][1], false, false);
        acc[1][0] = __builtin_amdgcn_wmma_f32_16x16x32_bf16(false, a1, false, b0, (short)0, acc[1][0], false, false);
        acc[1][1] = __builtin_amdgcn_wmma_f32_16x16x32_bf16(false, a1, false, b1, (short)0, acc[1][1], false, false);
    }

    int nl = lane & 15;
    int mh = (lane >> 4) << 3;
#pragma unroll
    for (int i = 0; i < 2; ++i) {
#pragma unroll
        for (int j = 0; j < 2; ++j) {
            int n  = n0 + j * 16 + nl;
            int mb = m0 + i * 16 + mh;
#pragma unroll
            for (int r = 0; r < 8; ++r) {
                float v = acc[i][j][r];
                long long idx = coff + (long long)(mb + r) * ldc + n;
                if (BF16OUT) ((u16*)Cout)[idx]   = f2bf(v);
                else         ((float*)Cout)[idx] = v;
            }
        }
    }
}

// ---------------------------------------------------------------------------
// LayerNorm over DIM=1024, one block per (padded) row, bf16 output; pad rows = 0.
// ---------------------------------------------------------------------------
__global__ __launch_bounds__(256)
void k_ln(const float* __restrict__ x, const float* __restrict__ g,
          const float* __restrict__ b, u16* __restrict__ out, int tokens)
{
    __shared__ float red[256];
    int row = blockIdx.x, tid = threadIdx.x;
    u16* orow = out + (long long)row * DIM;
    if (row >= tokens) {
        for (int i = tid; i < DIM; i += 256) orow[i] = 0;
        return;
    }
    const float* xr = x + (long long)row * DIM;
    float v[4], s = 0.f, ss = 0.f;
#pragma unroll
    for (int i = 0; i < 4; ++i) { v[i] = xr[tid + i * 256]; s += v[i]; ss += v[i] * v[i]; }
    red[tid] = s; __syncthreads();
    for (int o = 128; o > 0; o >>= 1) { if (tid < o) red[tid] += red[tid + o]; __syncthreads(); }
    float mean = red[0] * (1.f / DIM); __syncthreads();
    red[tid] = ss; __syncthreads();
    for (int o = 128; o > 0; o >>= 1) { if (tid < o) red[tid] += red[tid + o]; __syncthreads(); }
    float var = red[0] * (1.f / DIM) - mean * mean;
    float rs  = rsqrtf(var + 1e-6f);
#pragma unroll
    for (int i = 0; i < 4; ++i) {
        int c = tid + i * 256;
        orow[c] = f2bf((v[i] - mean) * rs * g[c] + b[c]);
    }
}

// ---------------------------------------------------------------------------
// QKV split: add biases, scale q by 1/sqrt(Dh), reshape to padded per-head
// layouts: q,k -> [BH, NPAD, DH] bf16 ; v -> transposed [BH, DH, NPAD] bf16.
// ---------------------------------------------------------------------------
__global__ __launch_bounds__(256)
void k_split(const float* __restrict__ qkv, const float* __restrict__ qb,
             const float* __restrict__ vb, u16* __restrict__ q,
             u16* __restrict__ k, u16* __restrict__ vT)
{
    long long idx = (long long)blockIdx.x * 256 + threadIdx.x;  // BH*NPAD*DH total
    int d = (int)(idx & (DH - 1));
    long long t1 = idx >> 6;
    int n  = (int)(t1 % NPAD);
    int bh = (int)(t1 / NPAD);
    int h = bh % HEADS, b = bh / HEADS;
    int c = h * DH + d;
    float qv = 0.f, kv = 0.f, vv = 0.f;
    if (n < SEQ) {
        const float* row = qkv + (long long)(b * SEQ + n) * (3 * DIM);
        qv = (row[c] + qb[c]) * 0.125f;          // 1/sqrt(64)
        kv =  row[DIM + c];
        vv =  row[2 * DIM + c] + vb[c];
    }
    q [((long long)bh * NPAD + n) * DH + d] = f2bf(qv);
    k [((long long)bh * NPAD + n) * DH + d] = f2bf(kv);
    vT[((long long)bh * DH + d) * NPAD + n] = f2bf(vv);
}

// ---------------------------------------------------------------------------
// Softmax row kernel: reads f32 scores row, adds rel_pos_bias, writes bf16
// probs IN PLACE over the f32 row (row-private; barriers separate rd/wr).
// ---------------------------------------------------------------------------
__global__ __launch_bounds__(256)
void k_softmax(float* __restrict__ scores, const float* __restrict__ relb)
{
    __shared__ float red[256];
    int row = blockIdx.x;              // [0, BH*NPAD)
    int qr = row % NPAD;
    int bh = row / NPAD;
    int h  = bh % HEADS;
    float* srow = scores + (long long)row * NPAD;
    u16*   prow = (u16*)srow;
    int tid = threadIdx.x;
    if (qr >= SEQ) {
        for (int i = tid; i < NPAD; i += 256) prow[i] = 0;
        return;
    }
    const float* brow = relb + ((long long)h * SEQ + qr) * SEQ;
    float v[3], mx = -3.4e38f;
#pragma unroll
    for (int i = 0; i < 3; ++i) {
        int j = tid + i * 256;
        float s = -3.4e38f;
        if (j < SEQ) s = srow[j] + brow[j];
        v[i] = s;
        mx = fmaxf(mx, s);
    }
    red[tid] = mx; __syncthreads();
    for (int o = 128; o > 0; o >>= 1) { if (tid < o) red[tid] = fmaxf(red[tid], red[tid + o]); __syncthreads(); }
    mx = red[0]; __syncthreads();
    float sum = 0.f;
#pragma unroll
    for (int i = 0; i < 3; ++i) {
        int j = tid + i * 256;
        float e = (j < SEQ) ? __expf(v[i] - mx) : 0.f;
        v[i] = e; sum += e;
    }
    red[tid] = sum; __syncthreads();
    for (int o = 128; o > 0; o >>= 1) { if (tid < o) red[tid] += red[tid + o]; __syncthreads(); }
    float inv = 1.f / red[0];
    __syncthreads();                   // all f32 reads done before bf16 overwrite
#pragma unroll
    for (int i = 0; i < 3; ++i) {
        int j = tid + i * 256;
        if (j < NPAD) prow[j] = (j < SEQ) ? f2bf(v[i] * inv) : 0;
    }
}

// ---------------------------------------------------------------------------
// Residual epilogues + weight conversion
// ---------------------------------------------------------------------------
__global__ __launch_bounds__(256)
void k_addproj(const float* __restrict__ x, const float* __restrict__ proj,
               const float* __restrict__ bproj, float* __restrict__ x1)
{
    long long idx = (long long)blockIdx.x * 256 + threadIdx.x;   // MTOK*DIM
    int c = (int)(idx & (DIM - 1));
    long long t = idx >> 10;
    int n = (int)(t % SEQ), b = (int)(t / SEQ);
    x1[idx] = x[idx] + proj[((long long)(b * NPAD + n)) * DIM + c] + bproj[c];
}

__global__ __launch_bounds__(256)
void k_final(const float* __restrict__ x1, const float* __restrict__ fc2,
             const float* __restrict__ bfc2, float* __restrict__ out)
{
    long long idx = (long long)blockIdx.x * 256 + threadIdx.x;   // MTOK*DIM
    int c = (int)(idx & (DIM - 1));
    out[idx] = x1[idx] + fc2[idx] + bfc2[c];
}

__global__ __launch_bounds__(256)
void k_cvt(const float* __restrict__ src, u16* __restrict__ dst, int n)
{
    int i = blockIdx.x * 256 + threadIdx.x;
    if (i < n) dst[i] = f2bf(src[i]);
}

// ---------------------------------------------------------------------------
extern "C" void kernel_launch(void* const* d_in, const int* in_sizes, int n_in,
                              void* d_out, int out_size, void* d_ws, size_t ws_size,
                              hipStream_t stream)
{
    const float* x     = (const float*)d_in[0];
    const float* relb  = (const float*)d_in[1];
    const float* wqkv  = (const float*)d_in[2];
    const float* qbias = (const float*)d_in[3];
    const float* vbias = (const float*)d_in[4];
    const float* wproj = (const float*)d_in[5];
    const float* bproj = (const float*)d_in[6];
    const float* ln1g  = (const float*)d_in[7];
    const float* ln1b  = (const float*)d_in[8];
    const float* ln2g  = (const float*)d_in[9];
    const float* ln2b  = (const float*)d_in[10];
    const float* wfc1  = (const float*)d_in[11];
    const float* bfc1  = (const float*)d_in[12];
    const float* wfc2  = (const float*)d_in[13];
    const float* bfc2  = (const float*)d_in[14];
    float* out = (float*)d_out;

    // ---- workspace bump allocator (256B aligned) ----
    char* p = (char*)d_ws;
    auto alloc = [&](size_t bytes) -> void* {
        void* r = (void*)p;
        p += (bytes + 255) & ~(size_t)255;
        return r;
    };
    u16*   wqkv_bf = (u16*)  alloc((size_t)3 * DIM * DIM * 2);
    u16*   wproj_bf= (u16*)  alloc((size_t)DIM * DIM * 2);
    u16*   wfc1_bf = (u16*)  alloc((size_t)FFN_DIM * DIM * 2);
    u16*   wfc2_bf = (u16*)  alloc((size_t)DIM * FFN_DIM * 2);
    u16*   h_bf    = (u16*)  alloc((size_t)MPAD * DIM * 2);          // ln1 then ln2
    u16*   q_bf    = (u16*)  alloc((size_t)BH * NPAD * DH * 2);
    u16*   k_bf    = (u16*)  alloc((size_t)BH * NPAD * DH * 2);
    u16*   vT_bf   = (u16*)  alloc((size_t)BH * DH * NPAD * 2);
    float* scores  = (float*)alloc((size_t)BH * NPAD * NPAD * 4);    // -> bf16 probs in place
    u16*   o_bf    = (u16*)  alloc((size_t)BATCH * NPAD * DIM * 2);  // merged heads
    float* proj_f  = (float*)alloc((size_t)BATCH * NPAD * DIM * 4);
    float* x1_f    = (float*)alloc((size_t)MTOK * DIM * 4);
    // qkv_f32 region reused for MLP intermediates (3*DIM*4 == FFN*2 + DIM*4)
    float* qkv_f   = (float*)alloc((size_t)MPAD * 3 * DIM * 4);
    u16*   m1_bf   = (u16*)qkv_f;
    float* fc2_f   = (float*)((char*)qkv_f + (size_t)MPAD * FFN_DIM * 2);

    // ---- weights -> bf16 ----
    k_cvt<<<(3*DIM*DIM)/256, 256, 0, stream>>>(wqkv, wqkv_bf, 3*DIM*DIM);
    k_cvt<<<(DIM*DIM)/256,   256, 0, stream>>>(wproj, wproj_bf, DIM*DIM);
    k_cvt<<<(FFN_DIM*DIM)/256, 256, 0, stream>>>(wfc1, wfc1_bf, FFN_DIM*DIM);
    k_cvt<<<(DIM*FFN_DIM)/256, 256, 0, stream>>>(wfc2, wfc2_bf, DIM*FFN_DIM);

    // ---- attention branch ----
    k_ln<<<MPAD, 256, 0, stream>>>(x, ln1g, ln1b, h_bf, MTOK);

    // qkv = h @ w_qkv^T : [MPAD, 3072]
    k_gemm_big<false,false><<<dim3(MPAD/64, (3*DIM)/256, 1), 256, 0, stream>>>(
        h_bf, wqkv_bf, nullptr, qkv_f, DIM, DIM, DIM, 3*DIM, 0, 0, 0);

    k_split<<<(BH*(long long)NPAD*DH)/256, 256, 0, stream>>>(qkv_f, qbias, vbias, q_bf, k_bf, vT_bf);

    // scores[z] = q[z] @ k[z]^T : z in [0, BH), [NPAD, NPAD]
    k_gemm_big<false,false><<<dim3(NPAD/64, NPAD/256, BH), 256, 0, stream>>>(
        q_bf, k_bf, nullptr, scores, DH, DH, DH, NPAD,
        (long long)NPAD*DH, (long long)NPAD*DH, (long long)NPAD*NPAD);

    k_softmax<<<BH*NPAD, 256, 0, stream>>>(scores, relb);

    // o[z] = probs[z] @ v[z] : probs is bf16 over f32 rows -> lda = 2*NPAD
    k_gemm_small<true><<<dim3(NPAD/128, DH/64, BH), 256, 0, stream>>>(
        (const u16*)scores, vT_bf, o_bf, NPAD, 2*NPAD, NPAD, DIM,
        (long long)2*NPAD*NPAD, (long long)DH*NPAD, HEADS,
        (long long)NPAD*DIM, (long long)DH);

    // proj = o @ w_proj^T : [B*NPAD, DIM]
    k_gemm_big<false,false><<<dim3((BATCH*NPAD)/64, DIM/256, 1), 256, 0, stream>>>(
        o_bf, wproj_bf, nullptr, proj_f, DIM, DIM, DIM, DIM, 0, 0, 0);

    k_addproj<<<(MTOK*DIM)/256, 256, 0, stream>>>(x, proj_f, bproj, x1_f);

    // ---- MLP branch ----
    k_ln<<<MPAD, 256, 0, stream>>>(x1_f, ln2g, ln2b, h_bf, MTOK);

    // m1 = gelu(h2 @ w_fc1^T + b_fc1) : [MPAD, 4096] bf16
    k_gemm_big<true,true><<<dim3(MPAD/64, FFN_DIM/256, 1), 256, 0, stream>>>(
        h_bf, wfc1_bf, bfc1, m1_bf, DIM, DIM, DIM, FFN_DIM, 0, 0, 0);

    // fc2 = m1 @ w_fc2^T : [MPAD, 1024] f32
    k_gemm_big<false,false><<<dim3(MPAD/64, DIM/256, 1), 256, 0, stream>>>(
        m1_bf, wfc2_bf, nullptr, fc2_f, FFN_DIM, FFN_DIM, FFN_DIM, DIM, 0, 0, 0);

    k_final<<<(MTOK*DIM)/256, 256, 0, stream>>>(x1_f, fc2_f, bfc2, out);
}
